// CBAGNN_Block_80685255623026
// MI455X (gfx1250) — compile-verified, matrix operand order checked by reading, not run on hardware
//
#include <hip/hip_runtime.h>
#include <hip/hip_bf16.h>

// ---------------- static problem dims (match reference) ----------------
constexpr int NN   = 131072;   // nodes
constexpr int GG   = 512;      // graphs
constexpr int NPGc = 256;      // nodes per graph
constexpr int EE   = 2097152;  // edges
constexpr int FINc = 64;
constexpr int FOUTc= 128;
constexpr int Rc   = 16;
constexpr int Kc   = 128;      // top-k per graph

typedef __attribute__((ext_vector_type(16))) _Float16 v16h;
typedef __attribute__((ext_vector_type(8)))  _Float16 v8h;
typedef __attribute__((ext_vector_type(8)))  float    v8f;
typedef __attribute__((ext_vector_type(4)))  unsigned int u32x4;
typedef __attribute__((ext_vector_type(8)))  int          i32x8;
typedef __attribute__((ext_vector_type(4)))  int          i32x4;

#if __has_builtin(__builtin_amdgcn_tensor_load_to_lds)
#define HAVE_TDM 1
#else
#define HAVE_TDM 0
#endif

// =====================================================================
// WMMA GEMM:  C[M,128] (f32) = A[M,K](f16,row) * Bt[128,K](f16, B^T)
//             (+ bias[n] if bias!=null, += C if accum)
// Block = 256 threads (8 waves) -> 128x128 macro-tile.
// The Bt weight panel (<=32KB) is staged into LDS once per block via the
// Tensor Data Mover (tensor_load_to_lds + s_wait_tensorcnt), then each
// wave computes a 16x128 strip: per 32-wide k-step one A fetch feeds
// eight v_wmma_f32_16x16x32_f16 issues (8x A reuse).
// =====================================================================
__global__ __launch_bounds__(256)
void wmma_gemm_kernel(const _Float16* __restrict__ A,
                      const _Float16* __restrict__ Bt,
                      float* __restrict__ C,
                      const float* __restrict__ bias,
                      int M, int K, int accum)
{
    __shared__ __align__(16) _Float16 sB[FOUTc * 128];   // up to K=128
    const int tot = FOUTc * K;                            // f16 elements

#if HAVE_TDM
    if (threadIdx.x < 32) {
        // ---- Tensor DMA descriptor (D#): flat 1-D tile, global -> LDS ----
        const unsigned lds_base = (unsigned)(uintptr_t)(void*)sB; // addr[31:0] == LDS offset
        const unsigned long long ga = (unsigned long long)(uintptr_t)Bt;
        u32x4 g0;
        g0[0] = 1u;                                    // count=1 (valid), user mode
        g0[1] = lds_base;                              // lds_addr
        g0[2] = (unsigned)ga;                          // global_addr[31:0]
        g0[3] = (unsigned)((ga >> 32) & 0x1FFFFFFu)    // global_addr[56:32]
              | (2u << 30);                            // type = 2 ("image")
        const unsigned td0 = (unsigned)tot;            // elements (data_size units)
        i32x8 g1;
        g1[0] = (int)(1u << 16);                       // data_size=1 (2 bytes); mask=0
        g1[1] = (int)((td0 & 0xFFFFu) << 16);          // tensor_dim0[15:0]
        g1[2] = (int)(((td0 >> 16) & 0xFFFFu)          // tensor_dim0[31:16]
              | (1u << 16));                           // tensor_dim1 = 1
        g1[3] = (int)((td0 & 0xFFFFu) << 16);          // tile_dim0 = tot (fits 16b)
        g1[4] = 1;                                     // tile_dim1 = 1, tile_dim2 = 0
        g1[5] = (int)td0;                              // tensor_dim0_stride[31:0]
        g1[6] = 0;                                     // stride0 hi, stride1 lo
        g1[7] = 0;                                     // stride1 hi
        i32x4 gz; gz[0]=0; gz[1]=0; gz[2]=0; gz[3]=0;
#if defined(__clang_major__) && (__clang_major__ >= 23)
        i32x8 gz8; for (int i = 0; i < 8; ++i) gz8[i] = 0;
        __builtin_amdgcn_tensor_load_to_lds(g0, g1, gz, gz, gz8, 0);
#else
        __builtin_amdgcn_tensor_load_to_lds(g0, g1, gz, gz, 0);
#endif
        __builtin_amdgcn_s_wait_tensorcnt(0);
    }
#else
    for (int i = threadIdx.x; i < (tot >> 3); i += 256)
        ((v8h*)sB)[i] = ((const v8h*)Bt)[i];
#endif
    __syncthreads();

    const int wave = threadIdx.x >> 5;
    const int lane = threadIdx.x & 31;
    const int half = lane >> 4;          // K-half selector per ISA A/B layout
    const int lr   = lane & 15;          // m for A, n for B/C
    const int m0   = blockIdx.x * 128 + wave * 16;

    v8f c[8];
    if (accum) {
#pragma unroll
        for (int nt = 0; nt < 8; ++nt)
#pragma unroll
            for (int v = 0; v < 8; ++v)
                c[nt][v] = C[(size_t)(m0 + v + 8 * half) * FOUTc + nt * 16 + lr];
    } else {
#pragma unroll
        for (int nt = 0; nt < 8; ++nt)
#pragma unroll
            for (int v = 0; v < 8; ++v) c[nt][v] = 0.0f;
    }

    const _Float16* arow = A + (size_t)(m0 + lr) * K;

    for (int kk = 0; kk < K; kk += 32) {
        union { v16h v; v8h h[2]; } ua;
        // A 16x32 f16 layout: elems 0..7  <- k = kk +      8*half + j
        //                     elems 8..15 <- k = kk + 16 + 8*half + j
        ua.h[0] = *(const v8h*)(arow + kk +      8 * half);
        ua.h[1] = *(const v8h*)(arow + kk + 16 + 8 * half);
#pragma unroll
        for (int nt = 0; nt < 8; ++nt) {
            union { v16h v; v8h h[2]; } ub;
            const _Float16* brow = sB + (size_t)(nt * 16 + lr) * K;
            ub.h[0] = *(const v8h*)(brow + kk +      8 * half);
            ub.h[1] = *(const v8h*)(brow + kk + 16 + 8 * half);
            c[nt] = __builtin_amdgcn_wmma_f32_16x16x32_f16(
                        false, ua.v, false, ub.v, (short)0, c[nt], false, false);
        }
    }

#pragma unroll
    for (int nt = 0; nt < 8; ++nt) {
        const float bv = bias ? bias[nt * 16 + lr] : 0.0f;
#pragma unroll
        for (int v = 0; v < 8; ++v)
            C[(size_t)(m0 + v + 8 * half) * FOUTc + nt * 16 + lr] = c[nt][v] + bv;
    }
}

// =====================================================================
// Graph LayerNorm (PyG mode='graph'): mean/var over all NPG*F elements
// of each graph; optional ELU; emits f32 and f16 copies.
// =====================================================================
__global__ __launch_bounds__(256)
void graph_norm_kernel(const float* __restrict__ in,
                       float* __restrict__ outf, _Float16* __restrict__ outh,
                       const float* __restrict__ w, const float* __restrict__ b,
                       int F, int apply_elu)
{
    __shared__ float ssum[256], ssq[256];
    const int g = blockIdx.x, t = threadIdx.x;
    const int total = NPGc * F;
    const float* base = in + (size_t)g * total;
    float s = 0.f, q = 0.f;
    for (int i = t; i < total; i += 256) { float v = base[i]; s += v; q += v * v; }
    ssum[t] = s; ssq[t] = q; __syncthreads();
    for (int o = 128; o > 0; o >>= 1) {
        if (t < o) { ssum[t] += ssum[t + o]; ssq[t] += ssq[t + o]; }
        __syncthreads();
    }
    const float mean = ssum[0] / (float)total;
    const float var  = ssq[0] / (float)total - mean * mean;
    const float rstd = rsqrtf(var + 1e-5f);
    for (int i = t; i < total; i += 256) {
        const int f = i % F;
        float v = (base[i] - mean) * rstd * w[f] + b[f];
        if (apply_elu) v = v > 0.f ? v : (expf(v) - 1.f);
        outf[(size_t)g * total + i] = v;
        outh[(size_t)g * total + i] = (_Float16)v;
    }
}

// Transpose-convert weights: in f32 [K,Nn] -> out f16 [Nn,K]
__global__ void convert_wT_kernel(const float* __restrict__ in,
                                  _Float16* __restrict__ out, int K, int Nn)
{
    int idx = blockIdx.x * 256 + threadIdx.x;
    if (idx >= K * Nn) return;
    int n = idx / K, k = idx % K;
    out[idx] = (_Float16)in[(size_t)k * Nn + n];
}

__global__ void fill_kernel(float* p, float v, int n)
{
    int i = blockIdx.x * 256 + threadIdx.x;
    if (i < n) p[i] = v;
}

// ---------------------------- RGCN ----------------------------
__global__ void rgcn_count_kernel(const int* __restrict__ dst,
                                  const int* __restrict__ et, float* cnt)
{
    int e = blockIdx.x * 256 + threadIdx.x;
    if (e >= EE) return;
    atomicAdd(&cnt[(size_t)dst[e] * Rc + et[e]], 1.0f);
}

__global__ void rgcn_scatter_kernel(const float* __restrict__ x,
                                    const int* __restrict__ src,
                                    const int* __restrict__ dst,
                                    const int* __restrict__ et,
                                    float* s_r, int r)
{
    int e = blockIdx.x * 256 + threadIdx.x;
    if (e >= EE) return;
    if (et[e] != r) return;
    const float* xs = x + (size_t)src[e] * FINc;
    float* sd = s_r + (size_t)dst[e] * FINc;
    __builtin_prefetch(xs, 0, 3);   // global_prefetch_b8
#pragma unroll 4
    for (int f = 0; f < FINc; ++f) atomicAdd(&sd[f], xs[f]);
}

__global__ void rgcn_div_kernel(const float* __restrict__ s_r,
                                const float* __restrict__ cnt,
                                _Float16* __restrict__ s_rh, int r)
{
    int idx = blockIdx.x * 256 + threadIdx.x;
    if (idx >= NN * FINc) return;
    int n = idx >> 6;
    float cd = fmaxf(cnt[(size_t)n * Rc + r], 1.0f);
    s_rh[idx] = (_Float16)(s_r[idx] / cd);
}

// ---------------------------- GAT ----------------------------
__global__ void gat_adot_kernel(const float* __restrict__ hh,
                                const float* __restrict__ a_s,
                                const float* __restrict__ a_d,
                                float* asrc, float* adst)
{
    int idx = blockIdx.x * 256 + threadIdx.x;
    if (idx >= NN * 2) return;
    int n = idx >> 1, hd = idx & 1;
    const float* hp = hh + (size_t)n * FOUTc + hd * 64;
    const float* as = a_s + hd * 64;
    const float* ad = a_d + hd * 64;
    float s1 = 0.f, s2 = 0.f;
    for (int f = 0; f < 64; ++f) { float v = hp[f]; s1 += v * as[f]; s2 += v * ad[f]; }
    asrc[idx] = s1; adst[idx] = s2;
}

__device__ inline void atomicMaxF(float* addr, float val)
{
    int old = __float_as_int(*addr);
    while (__int_as_float(old) < val) {
        int prev = atomicCAS((int*)addr, old, __float_as_int(val));
        if (prev == old) break;
        old = prev;
    }
}

__global__ void gat_max_kernel(const int* __restrict__ src, const int* __restrict__ dst,
                               const float* __restrict__ asrc, const float* __restrict__ adst,
                               float* emax)
{
    int idx = blockIdx.x * 256 + threadIdx.x;
    if (idx >= EE + NN) return;
    int s, d;
    if (idx < EE) { s = src[idx]; d = dst[idx]; } else { s = d = idx - EE; }
#pragma unroll
    for (int hd = 0; hd < 2; ++hd) {
        float e = asrc[s * 2 + hd] + adst[d * 2 + hd];
        e = e > 0.f ? e : 0.2f * e;
        atomicMaxF(&emax[d * 2 + hd], e);
    }
}

__global__ void gat_sum_kernel(const int* __restrict__ src, const int* __restrict__ dst,
                               const float* __restrict__ asrc, const float* __restrict__ adst,
                               const float* __restrict__ emax, float* denom)
{
    int idx = blockIdx.x * 256 + threadIdx.x;
    if (idx >= EE + NN) return;
    int s, d;
    if (idx < EE) { s = src[idx]; d = dst[idx]; } else { s = d = idx - EE; }
#pragma unroll
    for (int hd = 0; hd < 2; ++hd) {
        float e = asrc[s * 2 + hd] + adst[d * 2 + hd];
        e = e > 0.f ? e : 0.2f * e;
        atomicAdd(&denom[d * 2 + hd], expf(e - emax[d * 2 + hd]));
    }
}

__global__ void gat_agg_kernel(const int* __restrict__ src, const int* __restrict__ dst,
                               const float* __restrict__ asrc, const float* __restrict__ adst,
                               const float* __restrict__ emax, const float* __restrict__ denom,
                               const float* __restrict__ hh, float* out)
{
    int idx = blockIdx.x * 256 + threadIdx.x;
    if (idx >= (EE + NN) * 2) return;
    int ei = idx >> 1, hd = idx & 1;
    int s, d;
    if (ei < EE) { s = src[ei]; d = dst[ei]; } else { s = d = ei - EE; }
    float e = asrc[s * 2 + hd] + adst[d * 2 + hd];
    e = e > 0.f ? e : 0.2f * e;
    const float alpha = expf(e - emax[d * 2 + hd]) / (denom[d * 2 + hd] + 1e-16f);
    const float* hp = hh + (size_t)s * FOUTc + hd * 64;
    float* op = out + (size_t)d * FOUTc + hd * 64;
    for (int f = 0; f < 64; ++f) atomicAdd(&op[f], hp[f] * alpha);
}

__global__ void add_bias_kernel(float* h, const float* __restrict__ b)
{
    int i = blockIdx.x * 256 + threadIdx.x;
    if (i < NN * FOUTc) h[i] += b[i & (FOUTc - 1)];
}

// ---------------------------- pooling ----------------------------
__global__ void pool_agg_kernel(const int* __restrict__ src, const int* __restrict__ dst,
                                const float* __restrict__ h, float* agg)
{
    int idx = blockIdx.x * 256 + threadIdx.x;
    if (idx >= EE * 4) return;
    int e = idx >> 2, q = idx & 3;
    const float* hp = h + (size_t)src[e] * FOUTc + q * 32;
    float* ap = agg + (size_t)dst[e] * FOUTc + q * 32;
    for (int f = 0; f < 32; ++f) atomicAdd(&ap[f], hp[f]);
}

__global__ void score_kernel(const float* __restrict__ agg, const float* __restrict__ h,
                             const float* __restrict__ wprel, const float* __restrict__ wproot,
                             const float* __restrict__ bp, float* score)
{
    int n = blockIdx.x * 256 + threadIdx.x;
    if (n >= NN) return;
    float s = bp[0];
    const float* ap = agg + (size_t)n * FOUTc;
    const float* hp = h + (size_t)n * FOUTc;
    for (int f = 0; f < FOUTc; ++f) s += ap[f] * wprel[f] + hp[f] * wproot[f];
    score[n] = tanhf(s);
}

// Per-graph top-K via bitonic sort of 256 (value,index) pairs in LDS,
// descending (matches jax.lax.top_k ordering).
__global__ __launch_bounds__(256)
void topk_kernel(const float* __restrict__ score, const float* __restrict__ h,
                 float* att_x, float* att_batch)
{
    __shared__ float vals[256];
    __shared__ int   idxs[256];
    const int g = blockIdx.x, t = threadIdx.x;
    vals[t] = score[g * NPGc + t];
    idxs[t] = t;
    __syncthreads();
    for (int k = 2; k <= 256; k <<= 1) {
        for (int j = k >> 1; j > 0; j >>= 1) {
            int ixj = t ^ j;
            if (ixj > t) {
                bool up = ((t & k) == 0);           // 'up' segments: descending
                float v1 = vals[t], v2 = vals[ixj];
                bool sw = up ? (v1 < v2) : (v1 > v2);
                if (sw) {
                    vals[t] = v2; vals[ixj] = v1;
                    int i1 = idxs[t]; idxs[t] = idxs[ixj]; idxs[ixj] = i1;
                }
            }
            __syncthreads();
        }
    }
    if (t < Kc) {
        const float tv = vals[t];
        const float* hp = h + ((size_t)g * NPGc + idxs[t]) * FOUTc;
        float* op = att_x + ((size_t)g * Kc + t) * FOUTc;
        for (int f = 0; f < FOUTc; ++f) op[f] = hp[f] * tv;
        att_batch[g * Kc + t] = (float)g;
    }
}

__global__ __launch_bounds__(128)
void emb_kernel(const float* __restrict__ att_x, float* emb)
{
    const int g = blockIdx.x, f = threadIdx.x;
    float s = 0.f;
    for (int j = 0; j < Kc; ++j) s += att_x[((size_t)g * Kc + j) * FOUTc + f];
    emb[g * FOUTc + f] = s;
}

// =====================================================================
// host orchestration
// =====================================================================
static inline dim3 grid1(long long n) { return dim3((unsigned)((n + 255) / 256)); }

extern "C" void kernel_launch(void* const* d_in, const int* in_sizes, int n_in,
                              void* d_out, int out_size, void* d_ws, size_t ws_size,
                              hipStream_t stream)
{
    // -------- inputs (setup_inputs order) --------
    const float* x      = (const float*)d_in[0];
    const int*   src    = (const int*)d_in[1];
    const int*   dst    = (const int*)d_in[2];
    const int*   etype  = (const int*)d_in[3];
    /* batch d_in[4] implicit */
    const float* ln1w = (const float*)d_in[5],  *ln1b = (const float*)d_in[6];
    const float* ln2w = (const float*)d_in[7],  *ln2b = (const float*)d_in[8];
    const float* ln3w = (const float*)d_in[9],  *ln3b = (const float*)d_in[10];
    const float* W_rel  = (const float*)d_in[11];
    const float* W_root = (const float*)d_in[12];
    const float* rgcn_b = (const float*)d_in[13];
    const float* W2     = (const float*)d_in[14];
    const float* a2s    = (const float*)d_in[15], *a2d = (const float*)d_in[16];
    const float* b2     = (const float*)d_in[17];
    const float* W3     = (const float*)d_in[18];
    const float* a3s    = (const float*)d_in[19], *a3d = (const float*)d_in[20];
    const float* b3     = (const float*)d_in[21];
    const float* WpRoot = (const float*)d_in[22];
    const float* WpRel  = (const float*)d_in[23];
    const float* bp     = (const float*)d_in[24];

    // -------- workspace carve-out --------
    char* w = (char*)d_ws;
    float*     buf0  = (float*)w;      w += (size_t)NN * FOUTc * 4;
    float*     buf1  = (float*)w;      w += (size_t)NN * FOUTc * 4;
    _Float16*  buf0h = (_Float16*)w;   w += (size_t)NN * FOUTc * 2;
    float*     s_r   = (float*)w;      w += (size_t)NN * FINc * 4;
    _Float16*  s_rh  = (_Float16*)w;   w += (size_t)NN * FINc * 2;
    float*     cnt   = (float*)w;      w += (size_t)NN * Rc * 4;
    float*     asrc  = (float*)w;      w += (size_t)NN * 2 * 4;
    float*     adst  = (float*)w;      w += (size_t)NN * 2 * 4;
    float*     emax  = (float*)w;      w += (size_t)NN * 2 * 4;
    float*     denom = (float*)w;      w += (size_t)NN * 2 * 4;
    float*     score = (float*)w;      w += (size_t)NN * 4;
    _Float16*  wroot_t = (_Float16*)w; w += (size_t)FOUTc * FINc * 2;
    _Float16*  wrel_t  = (_Float16*)w; w += (size_t)Rc * FOUTc * FINc * 2;
    _Float16*  w2_t    = (_Float16*)w; w += (size_t)FOUTc * FOUTc * 2;
    _Float16*  w3_t    = (_Float16*)w; w += (size_t)FOUTc * FOUTc * 2;

    // -------- outputs --------
    float* out     = (float*)d_out;
    float* att_x   = out;                                   // [G*K, FOUT]
    float* att_b   = out + (size_t)GG * Kc * FOUTc;         // [G*K]
    float* emb_out = att_b + (size_t)GG * Kc;               // [G, FOUT]

    const dim3 gemmGrid(NN / 128);      // 8 waves/block, 128x128 macro-tile

    // ---- weight conversion (f32 -> f16, transposed for B operand) ----
    convert_wT_kernel<<<grid1(FINc * FOUTc), 256, 0, stream>>>(W_root, wroot_t, FINc, FOUTc);
    for (int r = 0; r < Rc; ++r)
        convert_wT_kernel<<<grid1(FINc * FOUTc), 256, 0, stream>>>(
            W_rel + (size_t)r * FINc * FOUTc, wrel_t + (size_t)r * FOUTc * FINc, FINc, FOUTc);
    convert_wT_kernel<<<grid1(FOUTc * FOUTc), 256, 0, stream>>>(W2, w2_t, FOUTc, FOUTc);
    convert_wT_kernel<<<grid1(FOUTc * FOUTc), 256, 0, stream>>>(W3, w3_t, FOUTc, FOUTc);

    // ---- 1. ln1: x -> buf0 (f32 [N,64]) + buf0h (f16 [N,64]) ----
    graph_norm_kernel<<<GG, 256, 0, stream>>>(x, buf0, buf0h, ln1w, ln1b, FINc, 0);

    // ---- 2. RGCN -> buf1 ----
    hipMemsetAsync(cnt, 0, (size_t)NN * Rc * 4, stream);
    rgcn_count_kernel<<<grid1(EE), 256, 0, stream>>>(dst, etype, cnt);
    // root term + bias
    wmma_gemm_kernel<<<gemmGrid, 256, 0, stream>>>(buf0h, wroot_t, buf1, rgcn_b,
                                                   NN, FINc, 0);
    // per-relation mean-aggregate + accumulating GEMM
    for (int r = 0; r < Rc; ++r) {
        hipMemsetAsync(s_r, 0, (size_t)NN * FINc * 4, stream);
        rgcn_scatter_kernel<<<grid1(EE), 256, 0, stream>>>(buf0, src, dst, etype, s_r, r);
        rgcn_div_kernel<<<grid1((long long)NN * FINc), 256, 0, stream>>>(s_r, cnt, s_rh, r);
        wmma_gemm_kernel<<<gemmGrid, 256, 0, stream>>>(s_rh, wrel_t + (size_t)r * FOUTc * FINc,
                                                       buf1, nullptr, NN, FINc, 1);
    }

    // ---- 3. ln2 + ELU: buf1 -> buf0/buf0h [N,128] ----
    graph_norm_kernel<<<GG, 256, 0, stream>>>(buf1, buf0, buf0h, ln2w, ln2b, FOUTc, 1);

    // ---- GAT layer helper (source features in buf0h) ----
    auto gat_layer = [&](const _Float16* Wt, const float* a_s, const float* a_d,
                         const float* bvec, float* hh, float* outb) {
        wmma_gemm_kernel<<<gemmGrid, 256, 0, stream>>>(buf0h, Wt, hh, nullptr,
                                                       NN, FOUTc, 0);
        gat_adot_kernel<<<grid1((long long)NN * 2), 256, 0, stream>>>(hh, a_s, a_d, asrc, adst);
        fill_kernel<<<grid1((long long)NN * 2), 256, 0, stream>>>(emax, -1e30f, NN * 2);
        hipMemsetAsync(denom, 0, (size_t)NN * 2 * 4, stream);
        gat_max_kernel<<<grid1((long long)EE + NN), 256, 0, stream>>>(src, dst, asrc, adst, emax);
        gat_sum_kernel<<<grid1((long long)EE + NN), 256, 0, stream>>>(src, dst, asrc, adst, emax, denom);
        hipMemsetAsync(outb, 0, (size_t)NN * FOUTc * 4, stream);
        gat_agg_kernel<<<grid1(((long long)EE + NN) * 2), 256, 0, stream>>>(
            src, dst, asrc, adst, emax, denom, hh, outb);
        add_bias_kernel<<<grid1((long long)NN * FOUTc), 256, 0, stream>>>(outb, bvec);
    };

    // ---- 4. GAT #1: hh in buf1, output in buf0 ----
    gat_layer(w2_t, a2s, a2d, b2, buf1, buf0);

    // ---- 5. ln3 + ELU: buf0 -> buf1/buf0h ----
    graph_norm_kernel<<<GG, 256, 0, stream>>>(buf0, buf1, buf0h, ln3w, ln3b, FOUTc, 1);

    // ---- 6. GAT #2: hh in buf0, output in buf1 (= final node features) ----
    gat_layer(w3_t, a3s, a3d, b3, buf0, buf1);

    // ---- 7. SAGPooling ----
    hipMemsetAsync(buf0, 0, (size_t)NN * FOUTc * 4, stream);
    pool_agg_kernel<<<grid1((long long)EE * 4), 256, 0, stream>>>(src, dst, buf1, buf0);
    score_kernel<<<grid1((long long)NN), 256, 0, stream>>>(buf0, buf1, WpRel, WpRoot, bp, score);
    topk_kernel<<<GG, 256, 0, stream>>>(score, buf1, att_x, att_b);
    emb_kernel<<<GG, 128, 0, stream>>>(att_x, emb_out);
}
